// TriangleAttention_6493990551931
// MI455X (gfx1250) — matrix operand order repaired and checked by reading, not compiled, not run on hardware
//
#include <hip/hip_runtime.h>
#include <hip/hip_bf16.h>
#include <math.h>

typedef _Float16 v16h __attribute__((ext_vector_type(16)));
typedef _Float16 v4h  __attribute__((ext_vector_type(4)));
typedef float    v8f  __attribute__((ext_vector_type(8)));
typedef float    v4f  __attribute__((ext_vector_type(4)));
typedef unsigned int v4u __attribute__((ext_vector_type(4)));

union AB16 { v4u q[2]; v16h v; };   // one WMMA f16 operand fragment (32 bytes)

#define LEN   256
#define CIN   128
#define HEADS 4
#define DIMD  32
#define NPOS  (LEN * LEN)           // 65536
#define INV_SQRT_D 0.17677669529663687f

static __device__ __forceinline__ v8f wmma_f16(const v16h& a, const v16h& b, const v8f& c) {
    return __builtin_amdgcn_wmma_f32_16x16x32_f16(false, a, false, b, (short)0, c, false, false);
}

// ---------------------------------------------------------------------------
// Kernel 0: pack weights into per-lane B-fragment layout (half).
// Wcat = [w_qkv (384 cols) | w_gate (128 cols)]  -> packQG [32nt][4kt][32lane][16e]
// w_out (128 cols)                               -> packO  [8nt][4kt][32lane][16e]
// ---------------------------------------------------------------------------
__global__ void ta_pack_w(const float* __restrict__ w_qkv,
                          const float* __restrict__ w_gate,
                          const float* __restrict__ w_out,
                          _Float16* __restrict__ packQG,
                          _Float16* __restrict__ packO) {
    int t = blockIdx.x * 128 + threadIdx.x;
    if (t < 32 * 4 * 32) {
        int lane = t & 31;
        int kt   = (t >> 5) & 3;
        int nt   = t >> 7;
        int col  = nt * 16 + (lane & 15);
        int kbase = kt * 32 + ((lane >> 4) << 4);
        for (int e = 0; e < 16; ++e) {
            int k = kbase + e;
            float v = (col < 384) ? w_qkv[k * 384 + col] : w_gate[k * 128 + (col - 384)];
            packQG[t * 16 + e] = (_Float16)v;
        }
    } else if (t < 32 * 4 * 32 + 8 * 4 * 32) {
        int t2 = t - 32 * 4 * 32;
        int lane = t2 & 31;
        int kt   = (t2 >> 5) & 3;
        int nt   = t2 >> 7;
        int col  = nt * 16 + (lane & 15);
        int kbase = kt * 32 + ((lane >> 4) << 4);
        for (int e = 0; e < 16; ++e)
            packO[t2 * 16 + e] = (_Float16)w_out[(kbase + e) * 128 + col];
    }
}

// ---------------------------------------------------------------------------
// Kernel 1: LayerNorm (one wave per position) -> zn (half)
//           bias = zn @ w_pair, stored TRANSPOSED: bias_t[h][pos]  (coalesced
//           reads in the attention kernel).
// ---------------------------------------------------------------------------
__global__ void ta_ln_bias(const float* __restrict__ z,
                           const float* __restrict__ gamma,
                           const float* __restrict__ beta,
                           const float* __restrict__ w_pair,
                           _Float16* __restrict__ zn_h,
                           float* __restrict__ bias_t) {
    int wid  = threadIdx.x >> 5;
    int lane = threadIdx.x & 31;
    int p = blockIdx.x * 8 + wid;

    v4f x = *(const v4f*)(z + (size_t)p * CIN + lane * 4);
    float s = x[0] + x[1] + x[2] + x[3];
    #pragma unroll
    for (int m = 1; m < 32; m <<= 1) s += __shfl_xor(s, m, 32);
    float mu = s * (1.0f / 128.0f);

    float xc[4];
    float s2 = 0.f;
    #pragma unroll
    for (int u = 0; u < 4; ++u) { xc[u] = x[u] - mu; s2 += xc[u] * xc[u]; }
    #pragma unroll
    for (int m = 1; m < 32; m <<= 1) s2 += __shfl_xor(s2, m, 32);
    float rs = rsqrtf(s2 * (1.0f / 128.0f) + 1e-5f);

    v4f g = *(const v4f*)(gamma + lane * 4);
    v4f b = *(const v4f*)(beta  + lane * 4);
    float zn[4];
    v4h outh;
    #pragma unroll
    for (int u = 0; u < 4; ++u) { zn[u] = xc[u] * rs * g[u] + b[u]; outh[u] = (_Float16)zn[u]; }
    *(v4h*)(zn_h + (size_t)p * CIN + lane * 4) = outh;

    float acc[4] = {0.f, 0.f, 0.f, 0.f};
    #pragma unroll
    for (int u = 0; u < 4; ++u) {
        int c = lane * 4 + u;
        v4f wp = *(const v4f*)(w_pair + c * 4);
        #pragma unroll
        for (int hh = 0; hh < 4; ++hh) acc[hh] += zn[u] * wp[hh];
    }
    #pragma unroll
    for (int hh = 0; hh < 4; ++hh)
        #pragma unroll
        for (int m = 1; m < 32; m <<= 1) acc[hh] += __shfl_xor(acc[hh], m, 32);
    if (lane == 0) {
        #pragma unroll
        for (int hh = 0; hh < 4; ++hh)
            bias_t[(size_t)hh * NPOS + p] = acc[hh];
    }
}

// ---------------------------------------------------------------------------
// Kernel 2: WMMA GEMM  [65536,128](half) x [128,512] -> qkv(half) + sigmoid gate(half)
// one wave per 16x16 output tile, K-loop of 4 (4 x v_wmma_f32_16x16x32_f16)
// ---------------------------------------------------------------------------
__global__ void ta_gemm_proj(const _Float16* __restrict__ zn,
                             const _Float16* __restrict__ pk,
                             const float* __restrict__ b_gate,
                             _Float16* __restrict__ qkv,
                             _Float16* __restrict__ gate) {
    int gid  = (blockIdx.x * 256 + threadIdx.x) >> 5;
    int lane = threadIdx.x & 31;
    int mt = gid >> 5;        // 4096 row tiles
    int nt = gid & 31;        // 32 col tiles (512 cols)
    int arow = mt * 16 + (lane & 15);
    int kb   = (lane >> 4) << 4;

    v8f c; for (int e = 0; e < 8; ++e) c[e] = 0.f;
    #pragma unroll
    for (int kt = 0; kt < 4; ++kt) {
        AB16 a, b;
        const _Float16* ap = zn + (size_t)arow * CIN + kt * 32 + kb;
        a.q[0] = *(const v4u*)(ap);
        a.q[1] = *(const v4u*)(ap + 8);
        const _Float16* bp = pk + ((size_t)(nt * 4 + kt) * 32 + lane) * 16;
        b.q[0] = *(const v4u*)(bp);
        b.q[1] = *(const v4u*)(bp + 8);
        c = wmma_f16(a.v, b.v, c);
    }
    int col = nt * 16 + (lane & 15);
    int rb  = mt * 16 + ((lane >> 4) << 3);
    if (col < 384) {
        #pragma unroll
        for (int r = 0; r < 8; ++r)
            qkv[(size_t)(rb + r) * 384 + col] = (_Float16)c[r];
    } else {
        int c2 = col - 384;
        float bg = b_gate[c2];
        #pragma unroll
        for (int r = 0; r < 8; ++r) {
            float xx = c[r] + bg;
            // sigmoid via hardware v_rcp_f32 (avoids div_scale/div_fmas expansion)
            gate[(size_t)(rb + r) * CIN + c2] =
                (_Float16)__builtin_amdgcn_rcpf(1.0f + __expf(-xx));
        }
    }
}

// ---------------------------------------------------------------------------
// Kernel 3: row attention. One block (4 waves) per (r, h).
// S = QK^T/sqrt(d) + bias, mask, softmax over j, O = P V, fold 1/rowsum + gate.
// ---------------------------------------------------------------------------
__global__ void ta_attn(const _Float16* __restrict__ qkv,
                        const float* __restrict__ bias_t,
                        const _Float16* __restrict__ gate,
                        const int* __restrict__ smask,
                        _Float16* __restrict__ attng) {
    __shared__ _Float16 Vt[DIMD][LEN];          // V transposed: [d][j]   (16 KB)
    __shared__ _Float16 Pm[4][16][LEN];         // probs per wave         (32 KB)

    int r = blockIdx.x >> 2;
    int h = blockIdx.x & 3;
    int tid = threadIdx.x, wid = tid >> 5, lane = tid & 31;
    int lr = lane & 15, lh = lane >> 4, kb = lh * 16;

    const float* biash = bias_t + (size_t)h * NPOS;   // [i][j] plane for this head

    // stage V^T in LDS
    for (int idx = tid; idx < DIMD * LEN; idx += 128) {
        int d = idx & 31, j = idx >> 5;
        Vt[d][j] = qkv[(size_t)(r * LEN + j) * 384 + 256 + h * DIMD + d];
    }
    __syncthreads();

    for (int it = wid; it < 16; it += 4) {      // uniform per wave: 4 i-tiles each
        // Q A-fragment (rows it*16..+15, K = d 0..31)
        AB16 a;
        {
            const _Float16* qp = qkv + (size_t)(r * LEN + it * 16 + lr) * 384 + h * DIMD;
            a.q[0] = *(const v4u*)(qp + kb);
            a.q[1] = *(const v4u*)(qp + kb + 8);
        }
        // S tiles: 16 x (16x16), K=32 -> one WMMA each
        float acc[16][8];
        #pragma unroll
        for (int jt = 0; jt < 16; ++jt) {
            AB16 bf;
            const _Float16* kp = qkv + (size_t)(r * LEN + jt * 16 + lr) * 384 + 128 + h * DIMD + lh * 16;
            bf.q[0] = *(const v4u*)(kp);
            bf.q[1] = *(const v4u*)(kp + 8);
            v8f cc; for (int e = 0; e < 8; ++e) cc[e] = 0.f;
            cc = wmma_f16(a.v, bf.v, cc);
            #pragma unroll
            for (int e = 0; e < 8; ++e) acc[jt][e] = cc[e];
        }
        // epilogue: scale + bias + mask   (logit rows: ibase+e, cols: jt*16+lr)
        int ibase = it * 16 + lh * 8;
        int si[8];
        #pragma unroll
        for (int e = 0; e < 8; ++e) si[e] = smask[ibase + e];
        #pragma unroll
        for (int jt = 0; jt < 16; ++jt) {
            int j = jt * 16 + lr;
            int sj = smask[j];
            #pragma unroll
            for (int e = 0; e < 8; ++e) {
                float bv = biash[(size_t)(ibase + e) * LEN + j];   // coalesced over lr
                bool msk = ((si[e] == 0) != (sj == 0));
                acc[jt][e] = msk ? -1e-9f : acc[jt][e] * INV_SQRT_D + bv;
            }
        }
        // row max (reduce over 16 cols in half-wave + 16 jt tiles)
        float rmax[8];
        #pragma unroll
        for (int e = 0; e < 8; ++e) rmax[e] = acc[0][e];
        #pragma unroll
        for (int jt = 1; jt < 16; ++jt)
            #pragma unroll
            for (int e = 0; e < 8; ++e) rmax[e] = fmaxf(rmax[e], acc[jt][e]);
        #pragma unroll
        for (int e = 0; e < 8; ++e)
            #pragma unroll
            for (int m = 1; m < 16; m <<= 1) rmax[e] = fmaxf(rmax[e], __shfl_xor(rmax[e], m, 32));
        // exp, row sum, write probs to LDS ([i_local][j] layout)
        float rsum[8];
        #pragma unroll
        for (int e = 0; e < 8; ++e) rsum[e] = 0.f;
        #pragma unroll
        for (int jt = 0; jt < 16; ++jt)
            #pragma unroll
            for (int e = 0; e < 8; ++e) {
                float pv = __expf(acc[jt][e] - rmax[e]);
                rsum[e] += pv;
                Pm[wid][lh * 8 + e][jt * 16 + lr] = (_Float16)pv;
            }
        #pragma unroll
        for (int e = 0; e < 8; ++e)
            #pragma unroll
            for (int m = 1; m < 16; m <<= 1) rsum[e] += __shfl_xor(rsum[e], m, 32);
        float rinv[8];
        #pragma unroll
        for (int e = 0; e < 8; ++e) rinv[e] = __builtin_amdgcn_rcpf(rsum[e]);

        asm volatile("s_wait_dscnt 0" ::: "memory");   // P writes -> P reads (same wave)

        // O = P @ V : N = 32 (2 tiles), K = 256 (8 x 32)
        #pragma unroll
        for (int nt = 0; nt < 2; ++nt) {
            v8f o; for (int e = 0; e < 8; ++e) o[e] = 0.f;
            #pragma unroll
            for (int kt = 0; kt < 8; ++kt) {
                AB16 pa, vb;
                const _Float16* pp = &Pm[wid][lr][kt * 32 + kb];
                pa.q[0] = *(const v4u*)(pp);
                pa.q[1] = *(const v4u*)(pp + 8);
                const _Float16* vp = &Vt[nt * 16 + lr][kt * 32 + lh * 16];
                vb.q[0] = *(const v4u*)(vp);
                vb.q[1] = *(const v4u*)(vp + 8);
                o = wmma_f16(pa.v, vb.v, o);
            }
            #pragma unroll
            for (int e = 0; e < 8; ++e) {
                size_t pos = (size_t)r * LEN + (ibase + e);   // output position (I=r, J=i_att)
                int ch = h * DIMD + nt * 16 + lr;
                float g = (float)gate[pos * CIN + ch];
                attng[pos * CIN + ch] = (_Float16)(o[e] * rinv[e] * g);
            }
        }
    }
}

// ---------------------------------------------------------------------------
// Kernel 4: WMMA GEMM  gated(half)[65536,128] x w_out[128,128] + b_out -> f32 out
// ---------------------------------------------------------------------------
__global__ void ta_gemm_out(const _Float16* __restrict__ ag,
                            const _Float16* __restrict__ pk,
                            const float* __restrict__ b_out,
                            float* __restrict__ out) {
    int gid  = (blockIdx.x * 256 + threadIdx.x) >> 5;
    int lane = threadIdx.x & 31;
    int mt = gid >> 3;        // 4096 row tiles
    int nt = gid & 7;         // 8 col tiles
    int arow = mt * 16 + (lane & 15);
    int kb   = (lane >> 4) << 4;

    v8f c; for (int e = 0; e < 8; ++e) c[e] = 0.f;
    #pragma unroll
    for (int kt = 0; kt < 4; ++kt) {
        AB16 a, b;
        const _Float16* ap = ag + (size_t)arow * CIN + kt * 32 + kb;
        a.q[0] = *(const v4u*)(ap);
        a.q[1] = *(const v4u*)(ap + 8);
        const _Float16* bp = pk + ((size_t)(nt * 4 + kt) * 32 + lane) * 16;
        b.q[0] = *(const v4u*)(bp);
        b.q[1] = *(const v4u*)(bp + 8);
        c = wmma_f16(a.v, b.v, c);
    }
    int col = nt * 16 + (lane & 15);
    int rb  = mt * 16 + ((lane >> 4) << 3);
    float bo = b_out[col];
    #pragma unroll
    for (int r = 0; r < 8; ++r)
        out[(size_t)(rb + r) * CIN + col] = c[r] + bo;
}

// ---------------------------------------------------------------------------
extern "C" void kernel_launch(void* const* d_in, const int* in_sizes, int n_in,
                              void* d_out, int out_size, void* d_ws, size_t ws_size,
                              hipStream_t stream) {
    const float* z       = (const float*)d_in[0];
    const float* gamma   = (const float*)d_in[1];
    const float* beta    = (const float*)d_in[2];
    const float* w_qkv   = (const float*)d_in[3];
    const float* w_pair  = (const float*)d_in[4];
    const float* w_gate  = (const float*)d_in[5];
    const float* b_gate  = (const float*)d_in[6];
    const float* w_out   = (const float*)d_in[7];
    const float* b_out   = (const float*)d_in[8];
    const int*   smask   = (const int*)d_in[9];
    float* out           = (float*)d_out;

    // workspace layout
    char* ws = (char*)d_ws;
    const size_t SZ_ZN   = (size_t)NPOS * CIN * 2;        // 16 MB  half
    const size_t SZ_QKV  = (size_t)NPOS * 384 * 2;        // 48 MB  half
    const size_t SZ_GATE = (size_t)NPOS * CIN * 2;        // 16 MB  half
    const size_t SZ_AG   = (size_t)NPOS * CIN * 2;        // 16 MB  half
    const size_t SZ_BIAS = (size_t)NPOS * 4 * 4;          //  1 MB  f32 (transposed [h][pos])
    const size_t SZ_PQG  = (size_t)32 * 4 * 32 * 16 * 2;  // 128 KB half
    const size_t SZ_PO   = (size_t)8 * 4 * 32 * 16 * 2;   //  32 KB half

    _Float16* zn_h   = (_Float16*)(ws);
    _Float16* qkv_h  = (_Float16*)(ws + SZ_ZN);
    _Float16* gate_h = (_Float16*)(ws + SZ_ZN + SZ_QKV);
    _Float16* ag_h   = (_Float16*)(ws + SZ_ZN + SZ_QKV + SZ_GATE);
    float*    bias_t = (float*)   (ws + SZ_ZN + SZ_QKV + SZ_GATE + SZ_AG);
    _Float16* packQG = (_Float16*)(ws + SZ_ZN + SZ_QKV + SZ_GATE + SZ_AG + SZ_BIAS);
    _Float16* packO  = (_Float16*)(ws + SZ_ZN + SZ_QKV + SZ_GATE + SZ_AG + SZ_BIAS + SZ_PQG);
    (void)ws_size; (void)in_sizes; (void)n_in; (void)out_size; (void)SZ_PO;

    ta_pack_w   <<<40,    128, 0, stream>>>(w_qkv, w_gate, w_out, packQG, packO);
    ta_ln_bias  <<<NPOS/8, 256, 0, stream>>>(z, gamma, beta, w_pair, zn_h, bias_t);
    ta_gemm_proj<<<16384, 256, 0, stream>>>(zn_h, packQG, b_gate, qkv_h, gate_h);
    ta_attn     <<<LEN*HEADS, 128, 0, stream>>>(qkv_h, bias_t, gate_h, smask, ag_h);
    ta_gemm_out <<<4096,  256, 0, stream>>>(ag_h, packO, b_out, out);
}
